// RNN_6244882448886
// MI455X (gfx1250) — compile-verified, matrix operand order checked by reading, not run on hardware
//
#include <hip/hip_runtime.h>
#include <hip/hip_bf16.h>

// ---------------------------------------------------------------------------
// LSTM caption decoder forward, MI455X (gfx1250, wave32, WMMA).
//
// Strategy: each wave owns a 16-row batch tile persistently across all 5
// timesteps (batch rows are independent in the recurrence). h/c live in the
// wave's private LDS slice, all matmuls are v_wmma_f32_16x16x32_f16 with f32
// accumulation. Weights are pre-converted to f16 into d_ws once per launch
// (~1 MB -> permanently L2 resident). No inter-wave synchronization needed.
// ---------------------------------------------------------------------------

typedef __attribute__((ext_vector_type(16))) _Float16 v16h;
typedef __attribute__((ext_vector_type(8)))  _Float16 v8h;
typedef __attribute__((ext_vector_type(8)))  float    v8f;

#define B_TOT    65536
#define T_STEPS  5          // T-1 = 6-1
#define CNN_DIM  512
#define HID      256
#define EMB      128
#define VOCAB    10
#define WAVES    8          // waves per block
#define ROWS     16         // batch rows per wave (one WMMA M tile)

// d_ws layout in _Float16 units
#define WS_WPROJ 0
#define WS_WIH   (HID * CNN_DIM)                 // 131072
#define WS_WHH   (WS_WIH + 4 * HID * EMB)        // 262144
#define WS_WFC   (WS_WHH + 4 * HID * HID)        // 524288  (padded to 16 rows)
#define WS_TOTAL (WS_WFC + 16 * HID)             // 528384 halves (~1.03 MB)

// per-wave LDS slice: [c f32 16x256 | h f16 2x16x256 | x f16 16x128]
#define LDS_C_OFF 0          // 16384 B (also reused as f16 16x512 feature stage)
#define LDS_H_OFF 16384      // 16384 B (double buffered)
#define LDS_X_OFF 32768      // 4096 B
#define LDS_PER_WAVE 36864   // 36 KB; x8 waves = 288 KB of the 320 KB WGP LDS

// ---------------------------------------------------------------------------
// One-time weight conversion f32 -> f16 into workspace (W_fc zero-padded).
// ---------------------------------------------------------------------------
__global__ void cvt_weights_kernel(const float* __restrict__ Wproj,
                                   const float* __restrict__ Wih,
                                   const float* __restrict__ Whh,
                                   const float* __restrict__ Wfc,
                                   _Float16* __restrict__ ws) {
    int i = blockIdx.x * blockDim.x + threadIdx.x;
    if (i >= WS_TOTAL) return;
    float v;
    if (i < WS_WIH) {
        v = Wproj[i];
    } else if (i < WS_WHH) {
        v = Wih[i - WS_WIH];
    } else if (i < WS_WFC) {
        v = Whh[i - WS_WHH];
    } else {
        int l = i - WS_WFC;
        int r = l / HID, c = l % HID;
        v = (r < VOCAB) ? Wfc[r * HID + c] : 0.0f;   // zero-pad rows 10..15
    }
    ws[i] = (_Float16)v;
}

// ---------------------------------------------------------------------------
// Fragment loader for a 16x32 f16 tile of a row-major matrix, laid out per the
// CDNA5 wave32 A-matrix convention (CDNA5 ISA §7.12.2):
//   lane l (0..15):  row = row0+l, K = k0+{0..7, 16..23}
//   lane l+16:       row = row0+l, K = k0+{8..15, 24..31}
// The same pattern serves the B operand: every matmul here is x @ W^T, so B's
// transpose IS the row-major W, loaded with this identical layout.
// Both 8-half groups are 16-byte aligned -> b128 loads (global or ds).
// ---------------------------------------------------------------------------
__device__ __forceinline__ v16h frag_ab(const _Float16* __restrict__ base,
                                        int ld, int row0, int k0,
                                        int lr, int hi) {
    const _Float16* p = base + (size_t)(row0 + lr) * ld + k0 + hi * 8;
    v8h a = *(const v8h*)(p);
    v8h b = *(const v8h*)(p + 16);
    return __builtin_shufflevector(a, b, 0, 1, 2, 3, 4, 5, 6, 7,
                                          8, 9, 10, 11, 12, 13, 14, 15);
}

__device__ __forceinline__ v8f wmma_f32(v16h a, v16h b, v8f c) {
    // (neg_a, A, neg_b, B, c_mod, C, reuse_a, reuse_b)
    return __builtin_amdgcn_wmma_f32_16x16x32_f16(false, a, false, b,
                                                  (short)0, c, false, false);
}

__device__ __forceinline__ float sigmf(float x) {
    return 1.0f / (1.0f + expf(-x));
}

__device__ __forceinline__ v8h pack8(float4 f0, float4 f1) {
    v8h h;
    h[0] = (_Float16)f0.x; h[1] = (_Float16)f0.y;
    h[2] = (_Float16)f0.z; h[3] = (_Float16)f0.w;
    h[4] = (_Float16)f1.x; h[5] = (_Float16)f1.y;
    h[6] = (_Float16)f1.z; h[7] = (_Float16)f1.w;
    return h;
}

// ---------------------------------------------------------------------------
// Main persistent-tile LSTM kernel. Block = 8 waves, each wave fully
// independent (own batch tile, own LDS slice) -> no barriers anywhere.
// ---------------------------------------------------------------------------
__global__ void lstm_caption_kernel(const float* __restrict__ features,
                                    const int*   __restrict__ captions,
                                    const float* __restrict__ emb,
                                    const float* __restrict__ b_proj,
                                    const float* __restrict__ b_ih,
                                    const float* __restrict__ b_hh,
                                    const float* __restrict__ b_fc,
                                    const _Float16* __restrict__ ws,
                                    float* __restrict__ out) {
    extern __shared__ char smem[];
    const int lane = threadIdx.x & 31;
    const int wave = threadIdx.x >> 5;
    const int lr   = lane & 15;   // A/B row within tile; C/D column
    const int hi   = lane >> 4;   // half-wave select
    const int b0   = (blockIdx.x * WAVES + wave) * ROWS;

    char* wbase = smem + (size_t)wave * LDS_PER_WAVE;
    float*    cbuf = (float*)(wbase + LDS_C_OFF);      // [16][256] f32 cell state
    _Float16* fstg = (_Float16*)(wbase + LDS_C_OFF);   // feature stage view [16][512] f16
    _Float16* hbuf = (_Float16*)(wbase + LDS_H_OFF);   // 2 x [16][256] f16 (ping-pong)
    _Float16* xbuf = (_Float16*)(wbase + LDS_X_OFF);   // [16][128] f16 embeddings

    const _Float16* wProj = ws + WS_WPROJ;
    const _Float16* wIh   = ws + WS_WIH;
    const _Float16* wHh   = ws + WS_WHH;
    const _Float16* wFc   = ws + WS_WFC;

    // ---- stage this wave's 16x512 feature tile into LDS as f16 ------------
    {
        const float* src = features + (size_t)b0 * CNN_DIM;
        for (int i = lane * 8; i < ROWS * CNN_DIM; i += 32 * 8) {
            float4 f0 = *(const float4*)(src + i);
            float4 f1 = *(const float4*)(src + i + 4);
            *(v8h*)(fstg + i) = pack8(f0, f1);
        }
    }

    // ---- h0 = features @ Wproj^T + b_proj  (K = 512 -> 16 WMMAs/tile) -----
    _Float16* h0 = hbuf;  // buffer 0
    for (int n = 0; n < HID / 16; ++n) {
        v8f acc = {};
        #pragma unroll
        for (int k = 0; k < CNN_DIM / 32; ++k) {
            v16h A  = frag_ab(fstg,  CNN_DIM, 0,      k * 32, lr, hi);
            v16h Bw = frag_ab(wProj, CNN_DIM, n * 16, k * 32, lr, hi);
            acc = wmma_f32(A, Bw, acc);
        }
        float bp = b_proj[n * 16 + lr];
        #pragma unroll
        for (int r = 0; r < 8; ++r)
            h0[(r + 8 * hi) * HID + n * 16 + lr] = (_Float16)(acc[r] + bp);
    }

    // ---- c0 = 0 (feature stage no longer needed) --------------------------
    for (int i = lane * 4; i < ROWS * HID; i += 32 * 4)
        *(float4*)(cbuf + i) = make_float4(0.f, 0.f, 0.f, 0.f);

    // ---- 5 recurrent steps ------------------------------------------------
    for (int t = 0; t < T_STEPS; ++t) {
        _Float16* hprev = hbuf + (t & 1) * (ROWS * HID);
        _Float16* hnext = hbuf + ((t + 1) & 1) * (ROWS * HID);

        // gather x_t = emb[captions[:, t]] into LDS as f16
        {
            int row = lane >> 1;
            int c0  = (lane & 1) * 64;
            int tok = captions[(size_t)(b0 + row) * 6 + t];
            const float* e = emb + (size_t)tok * EMB + c0;
            _Float16* xd = xbuf + row * EMB + c0;
            #pragma unroll
            for (int j = 0; j < 64; j += 8) {
                float4 f0 = *(const float4*)(e + j);
                float4 f1 = *(const float4*)(e + j + 4);
                *(v8h*)(xd + j) = pack8(f0, f1);
            }
        }

        // gates: 16 column-tiles of h; i/f/g/o computed together so the
        // A fragment is loaded once per K-chunk and reused 4x.
        for (int n = 0; n < HID / 16; ++n) {
            v8f ai = {}, af = {}, ag = {}, ao = {};
            #pragma unroll
            for (int k = 0; k < EMB / 32; ++k) {        // x @ W_ih^T
                v16h A = frag_ab(xbuf, EMB, 0, k * 32, lr, hi);
                ai = wmma_f32(A, frag_ab(wIh, EMB, 0 * HID + n * 16, k * 32, lr, hi), ai);
                af = wmma_f32(A, frag_ab(wIh, EMB, 1 * HID + n * 16, k * 32, lr, hi), af);
                ag = wmma_f32(A, frag_ab(wIh, EMB, 2 * HID + n * 16, k * 32, lr, hi), ag);
                ao = wmma_f32(A, frag_ab(wIh, EMB, 3 * HID + n * 16, k * 32, lr, hi), ao);
            }
            #pragma unroll
            for (int k = 0; k < HID / 32; ++k) {        // h @ W_hh^T
                v16h A = frag_ab(hprev, HID, 0, k * 32, lr, hi);
                ai = wmma_f32(A, frag_ab(wHh, HID, 0 * HID + n * 16, k * 32, lr, hi), ai);
                af = wmma_f32(A, frag_ab(wHh, HID, 1 * HID + n * 16, k * 32, lr, hi), af);
                ag = wmma_f32(A, frag_ab(wHh, HID, 2 * HID + n * 16, k * 32, lr, hi), ag);
                ao = wmma_f32(A, frag_ab(wHh, HID, 3 * HID + n * 16, k * 32, lr, hi), ao);
            }

            const int gcol = n * 16 + lr;
            float bi_ = b_ih[0 * HID + gcol] + b_hh[0 * HID + gcol];
            float bf_ = b_ih[1 * HID + gcol] + b_hh[1 * HID + gcol];
            float bg_ = b_ih[2 * HID + gcol] + b_hh[2 * HID + gcol];
            float bo_ = b_ih[3 * HID + gcol] + b_hh[3 * HID + gcol];

            #pragma unroll
            for (int r = 0; r < 8; ++r) {
                float iv = sigmf(ai[r] + bi_);
                float fv = sigmf(af[r] + bf_);
                float gv = tanhf(ag[r] + bg_);
                float ov = sigmf(ao[r] + bo_);
                int idx = (r + 8 * hi) * HID + gcol;
                float cn = fv * cbuf[idx] + iv * gv;
                cbuf[idx]  = cn;
                hnext[idx] = (_Float16)(ov * tanhf(cn));
            }
        }

        // logits = h_next @ W_fc^T + b_fc  (one 16-wide N tile, cols 0..9)
        v8f acc = {};
        #pragma unroll
        for (int k = 0; k < HID / 32; ++k) {
            v16h A = frag_ab(hnext, HID, 0, k * 32, lr, hi);
            acc = wmma_f32(A, frag_ab(wFc, HID, 0, k * 32, lr, hi), acc);
        }
        if (lr < VOCAB) {
            float bb = b_fc[lr];
            #pragma unroll
            for (int r = 0; r < 8; ++r) {
                int row = b0 + r + 8 * hi;
                out[((size_t)row * T_STEPS + t) * VOCAB + lr] = acc[r] + bb;
            }
        }
    }
}

// ---------------------------------------------------------------------------
extern "C" void kernel_launch(void* const* d_in, const int* in_sizes, int n_in,
                              void* d_out, int out_size, void* d_ws, size_t ws_size,
                              hipStream_t stream) {
    (void)in_sizes; (void)n_in; (void)out_size; (void)ws_size;

    const float* features = (const float*)d_in[0];
    const int*   captions = (const int*)d_in[1];
    // d_in[2] = teacher_forcing_ratio (== 1, unused: reference always consumes captions)
    const float* emb   = (const float*)d_in[3];
    const float* Wproj = (const float*)d_in[4];
    const float* bproj = (const float*)d_in[5];
    const float* Wih   = (const float*)d_in[6];
    const float* Whh   = (const float*)d_in[7];
    const float* bih   = (const float*)d_in[8];
    const float* bhh   = (const float*)d_in[9];
    const float* Wfc   = (const float*)d_in[10];
    const float* bfc   = (const float*)d_in[11];
    _Float16* ws  = (_Float16*)d_ws;   // needs ~1.03 MB of workspace
    float*    out = (float*)d_out;

    cvt_weights_kernel<<<(WS_TOTAL + 255) / 256, 256, 0, stream>>>(
        Wproj, Wih, Whh, Wfc, ws);

    dim3 grid(B_TOT / (ROWS * WAVES));          // 512 blocks
    dim3 block(WAVES * 32);                     // 256 threads = 8 waves
    lstm_caption_kernel<<<grid, block, WAVES * LDS_PER_WAVE, stream>>>(
        features, captions, emb, bproj, bih, bhh, bfc, ws, out);
}
